// OrkyTransformer_18537078850000
// MI455X (gfx1250) — matrix operand order repaired and checked by reading, not compiled, not run on hardware
//
#include <hip/hip_runtime.h>
#include <hip/hip_bf16.h>

// ---------------- Problem constants (from reference) ----------------
#define V_  32000
#define D_  1024
#define H_  16
#define L_  2
#define F_  4096
#define S_  1024
#define B_  2
#define HD_ 64
#define BS_ (B_ * S_)
#define EPS_ 1e-6f

// CDNA5 wave32 WMMA fragment types
typedef __attribute__((ext_vector_type(16))) __bf16 v16bf;
typedef __attribute__((ext_vector_type(8)))  float  v8f;

// GEMM tiling: 256 threads = 8 wave32 (4x2 wave grid), block tile 128(M) x 64(N).
// K step 64 = two WMMA k-slices per LDS stage -> 8 v_wmma per barrier per wave.
// Double-buffered LDS, software-pipelined: stage k+1 while computing k.
#define TM 128
#define TN 64
#define TK 64
#define LDAS 72   // LDS row stride (bf16): 144B rows -> 16B-aligned b128, conflict-free frags
#define LDBS 72

// CDNA5 async copy global->LDS (ASYNCcnt). Flip to 0 if the assembler rejects it.
#define USE_ASYNC_COPY 1

__device__ __forceinline__ void async_b128(unsigned lds_addr, const __bf16* g) {
#if USE_ASYNC_COPY
  asm volatile("global_load_async_to_lds_b128 %0, %1, off"
               :: "v"(lds_addr), "v"((unsigned long long)(size_t)g)
               : "memory");
#endif
}
__device__ __forceinline__ void wait_async0() {
#if USE_ASYNC_COPY
  asm volatile("s_wait_asynccnt 0x0" ::: "memory");
#endif
}

enum {
  GF_BIAS = 1, GF_RELU = 2, GF_RESID = 4, GF_WF32 = 8, GF_WBF16 = 16, GF_BTRANS = 32
};

struct GemmArgs {
  const __bf16* A;  long long sA1, sA2; int lda;   // batch offset = (z/zdiv)*s1 + (z%zdiv)*s2
  const __bf16* Bm; long long sB1, sB2; int ldb;
  float* Cf; __bf16* Cb; long long sC1, sC2; int ldc;
  const float* bias;                                // per output column
  const float* resid; long long sR1, sR2;           // same ldc layout as C
  int M, N, K, zdiv;
};

union FragBF { uint4 u[2]; v16bf v; };
union Pack8  { uint4 u; __bf16 h[8]; };

template <int FLAGS>
__global__ __launch_bounds__(256)
void gemm_bf16_wmma_kernel(GemmArgs a) {
  __shared__ __align__(16) __bf16 lA[2][TM * LDAS];
  __shared__ __align__(16) __bf16 lB[2][TN * LDBS];

  const int tid   = threadIdx.x;
  const int lane  = tid & 31;
  const int wave  = tid >> 5;
  const int waveM = wave & 3;      // rows waveM*32
  const int waveN = wave >> 2;     // cols waveN*32
  const int half  = lane >> 4;
  const int l16   = lane & 15;

  const int z  = blockIdx.z;
  const int zq = z / a.zdiv, zr = z % a.zdiv;
  const __bf16* A  = a.A  + zq * a.sA1 + zr * a.sA2;
  const __bf16* Bm = a.Bm + zq * a.sB1 + zr * a.sB2;
  const long long coff = zq * a.sC1 + zr * a.sC2;
  const long long roff = (FLAGS & GF_RESID) ? (zq * a.sR1 + zr * a.sR2) : 0;
  const int m0 = blockIdx.y * TM;
  const int n0 = blockIdx.x * TN;

  // ---- staging geometry: A 128x64 -> 32 elems/thread; B 64x64 -> 16 elems/thread ----
  const int rA = tid >> 1;             // 0..127
  const int cA = (tid & 1) * 32;       // 0 / 32
  const int bI = tid >> 2;             // trans: n row (0..63); normal: k row (0..63)
  const int bJ = (tid & 3) * 16;       // trans: k base;        normal: n base

  auto gA_ptr = [&](int kb) { return A + (long long)(m0 + rA) * a.lda + kb + cA; };

  auto stageA_async = [&](int kb, int buf) {
    const __bf16* g = gA_ptr(kb);
    const unsigned l = (unsigned)(size_t)&lA[buf][rA * LDAS + cA];
#pragma unroll
    for (int j = 0; j < 4; ++j) async_b128(l + j * 16, g + j * 8);
  };
  auto loadA = [&](int kb, uint4* ra) {
    const __bf16* g = gA_ptr(kb);
#pragma unroll
    for (int j = 0; j < 4; ++j) ra[j] = *(const uint4*)(g + j * 8);
  };
  auto storeA = [&](int buf, const uint4* ra) {
#pragma unroll
    for (int j = 0; j < 4; ++j) *(uint4*)&lA[buf][rA * LDAS + cA + j * 8] = ra[j];
  };
  auto loadB = [&](int kb, uint4* rb) {
    if (FLAGS & GF_BTRANS) {   // Bsrc is [N][K] row-major: contiguous in k
      const __bf16* g = Bm + (long long)(n0 + bI) * a.ldb + kb + bJ;
      rb[0] = *(const uint4*)g;
      rb[1] = *(const uint4*)(g + 8);
    } else {                   // Bsrc is [K][N] row-major: coalesced along n
      const __bf16* g = Bm + (long long)(kb + bI) * a.ldb + n0 + bJ;
      rb[0] = *(const uint4*)g;
      rb[1] = *(const uint4*)(g + 8);
    }
  };
  auto storeB = [&](int buf, const uint4* rb) {   // LDS holds B^T (N-major)
    if (FLAGS & GF_BTRANS) {
      *(uint4*)&lB[buf][bI * LDBS + bJ]     = rb[0];
      *(uint4*)&lB[buf][bI * LDBS + bJ + 8] = rb[1];
    } else {
      Pack8 p0, p1; p0.u = rb[0]; p1.u = rb[1];
#pragma unroll
      for (int i = 0; i < 8; ++i) {
        lB[buf][(bJ + i)     * LDBS + bI] = p0.h[i];
        lB[buf][(bJ + 8 + i) * LDBS + bI] = p1.h[i];
      }
    }
  };

  v8f acc[2][2];
  const v8f vzero = {0.f, 0.f, 0.f, 0.f, 0.f, 0.f, 0.f, 0.f};
  acc[0][0] = vzero; acc[0][1] = vzero; acc[1][0] = vzero; acc[1][1] = vzero;

  // A frag (ISA 7.12.2): lanes 0-15 row l16 carry K 0-7 & 16-23; lanes 16-31 K 8-15 & 24-31.
  // B frag: column l16, same K split as A (half*8..+7 & 16+half*8..+7) from the N-major LDS tile.
  auto compute = [&](int buf) {
#pragma unroll
    for (int ks = 0; ks < 2; ++ks) {
      const int ko = ks * 32;
      FragBF fa[2], fb[2];
#pragma unroll
      for (int t = 0; t < 2; ++t) {
        const __bf16* pa = &lA[buf][(waveM * 32 + t * 16 + l16) * LDAS + ko + half * 8];
        fa[t].u[0] = *(const uint4*)pa;
        fa[t].u[1] = *(const uint4*)(pa + 16);
        const __bf16* pb = &lB[buf][(waveN * 32 + t * 16 + l16) * LDBS + ko + half * 8];
        fb[t].u[0] = *(const uint4*)pb;
        fb[t].u[1] = *(const uint4*)(pb + 16);
      }
#pragma unroll
      for (int ti = 0; ti < 2; ++ti)
#pragma unroll
        for (int tj = 0; tj < 2; ++tj)
          acc[ti][tj] = __builtin_amdgcn_wmma_f32_16x16x32_bf16(
              false, fa[ti].v, false, fb[tj].v, (short)0, acc[ti][tj], false, false);
    }
  };

  // ---- prologue: stage tile 0 ----
  int cur = 0;
  {
#if USE_ASYNC_COPY
    stageA_async(0, 0);
#else
    uint4 ra0[4]; loadA(0, ra0); storeA(0, ra0);
#endif
    uint4 rb0[2]; loadB(0, rb0); storeB(0, rb0);
    wait_async0();
    __syncthreads();
  }

  // ---- pipelined main loop: one barrier per K tile ----
  for (int kb = 0; kb < a.K; kb += TK) {
    const bool nxt = (kb + TK) < a.K;
    uint4 ra[4]; uint4 rb[2];
    (void)ra;
    if (nxt) {
#if USE_ASYNC_COPY
      stageA_async(kb + TK, cur ^ 1);        // DMA overlaps the WMMAs below
#else
      loadA(kb + TK, ra);
#endif
      loadB(kb + TK, rb);
      if (kb + 2 * TK < a.K) __builtin_prefetch(gA_ptr(kb + 2 * TK), 0, 1);
    }
    compute(cur);
    if (nxt) {
#if !USE_ASYNC_COPY
      storeA(cur ^ 1, ra);
#endif
      storeB(cur ^ 1, rb);
      wait_async0();
      __syncthreads();
      cur ^= 1;
    }
  }

  // ---- epilogue: c[r] -> row r + 8*(lane>=16), col = lane&15 ----
#pragma unroll
  for (int ti = 0; ti < 2; ++ti) {
#pragma unroll
    for (int tj = 0; tj < 2; ++tj) {
      const int col  = n0 + waveN * 32 + tj * 16 + l16;
      const int rowb = m0 + waveM * 32 + ti * 16 + half * 8;
      const float bc = (FLAGS & GF_BIAS) ? a.bias[col] : 0.f;
#pragma unroll
      for (int r = 0; r < 8; ++r) {
        float vv = acc[ti][tj][r] + bc;
        if (FLAGS & GF_RELU) vv = fmaxf(vv, 0.f);
        const long long rc = (long long)(rowb + r) * a.ldc + col;
        if (FLAGS & GF_RESID) vv += a.resid[roff + rc];
        if (FLAGS & GF_WF32)  a.Cf[coff + rc] = vv;
        if (FLAGS & GF_WBF16) a.Cb[coff + rc] = (__bf16)vv;
      }
    }
  }
}

// x = emb[tok] + pos, in f32 (residual path) and bf16 (WMMA A operand)
__global__ void embed_kernel(const int* __restrict__ tok, const float* __restrict__ emb,
                             const float* __restrict__ pos, float* __restrict__ xf,
                             __bf16* __restrict__ xb) {
  const long long i = (long long)blockIdx.x * 256 + threadIdx.x;
  const int d   = (int)(i & (D_ - 1));
  const int row = (int)(i >> 10);
  const int s   = row & (S_ - 1);
  const float v = emb[(long long)tok[row] * D_ + d] + pos[s * D_ + d];
  xf[i] = v;
  xb[i] = (__bf16)v;
}

__global__ void cvt_bf16_kernel(const float* __restrict__ src, __bf16* __restrict__ dst,
                                long long n) {
  const long long i = (long long)blockIdx.x * 256 + threadIdx.x;
  if (i < n) dst[i] = (__bf16)src[i];
}

// [H][D][HD] f32 -> [D][H*HD] bf16 so QKV becomes one [BS,D]x[D,D] GEMM
__global__ void pack_qkv_kernel(const float* __restrict__ src, __bf16* __restrict__ dst) {
  const long long i = (long long)blockIdx.x * 256 + threadIdx.x;
  const int e = (int)(i % HD_);
  const long long r = i / HD_;
  const int d = (int)(r % D_);
  const int h = (int)(r / D_);
  dst[(long long)d * D_ + h * HD_ + e] = (__bf16)src[i];
}

// one 1024-wide softmax row per block; scale = 1/sqrt(HD)
__global__ __launch_bounds__(256)
void softmax_bf16_kernel(const __bf16* __restrict__ in, __bf16* __restrict__ out, float scale) {
  __shared__ float red[256];
  const long long row = blockIdx.x;
  const __bf16* rp = in + row * S_;
  __bf16* wp = out + row * S_;
  const int t = threadIdx.x;
  float x[4];
#pragma unroll
  for (int i = 0; i < 4; ++i) x[i] = (float)rp[t + i * 256] * scale;
  float mx = fmaxf(fmaxf(x[0], x[1]), fmaxf(x[2], x[3]));
  red[t] = mx; __syncthreads();
  for (int s2 = 128; s2 > 0; s2 >>= 1) { if (t < s2) red[t] = fmaxf(red[t], red[t + s2]); __syncthreads(); }
  mx = red[0]; __syncthreads();
  float sum = 0.f;
#pragma unroll
  for (int i = 0; i < 4; ++i) { x[i] = __expf(x[i] - mx); sum += x[i]; }
  red[t] = sum; __syncthreads();
  for (int s2 = 128; s2 > 0; s2 >>= 1) { if (t < s2) red[t] += red[t + s2]; __syncthreads(); }
  const float inv = 1.f / red[0];
#pragma unroll
  for (int i = 0; i < 4; ++i) wp[t + i * 256] = (__bf16)(x[i] * inv);
}

// OrkLayerNorm: unbiased var (ddof=1), eps added to std; dual f32/bf16 output
__global__ __launch_bounds__(256)
void layernorm_kernel(const float* __restrict__ in, const float* __restrict__ g,
                      const float* __restrict__ b, float* __restrict__ of,
                      __bf16* __restrict__ ob) {
  __shared__ float r1[256], r2[256];
  const long long row = blockIdx.x;
  const float* x = in + row * D_;
  const int t = threadIdx.x;
  float v[4]; float s = 0.f, q = 0.f;
#pragma unroll
  for (int i = 0; i < 4; ++i) { v[i] = x[t + i * 256]; s += v[i]; q += v[i] * v[i]; }
  r1[t] = s; r2[t] = q; __syncthreads();
  for (int s2 = 128; s2 > 0; s2 >>= 1) {
    if (t < s2) { r1[t] += r1[t + s2]; r2[t] += r2[t + s2]; }
    __syncthreads();
  }
  const float mean = r1[0] / D_;
  const float var  = (r2[0] - D_ * mean * mean) / (float)(D_ - 1);
  const float inv  = 1.f / (sqrtf(var) + EPS_);
#pragma unroll
  for (int i = 0; i < 4; ++i) {
    const int d = t + i * 256;
    const float y = g[d] * (v[i] - mean) * inv + b[d];
    of[row * D_ + d] = y;
    ob[row * D_ + d] = (__bf16)y;
  }
}

static void launch_gemm(hipStream_t stream, const GemmArgs& ga, int nz, int flags) {
  dim3 grid(ga.N / TN, ga.M / TM, nz);
#define GEMM_CASE(F) case (F): gemm_bf16_wmma_kernel<(F)><<<grid, 256, 0, stream>>>(ga); break;
  switch (flags) {
    GEMM_CASE(GF_BIAS | GF_WBF16)
    GEMM_CASE(GF_WBF16 | GF_BTRANS)
    GEMM_CASE(GF_WBF16)
    GEMM_CASE(GF_BIAS | GF_RESID | GF_WF32)
    GEMM_CASE(GF_BIAS | GF_RELU | GF_WBF16)
    GEMM_CASE(GF_BIAS | GF_WF32)
    default: break;
  }
#undef GEMM_CASE
}

extern "C" void kernel_launch(void* const* d_in, const int* in_sizes, int n_in,
                              void* d_out, int out_size, void* d_ws, size_t ws_size,
                              hipStream_t stream) {
  (void)in_sizes; (void)n_in; (void)out_size; (void)ws_size;
  const int*   tokens = (const int*)d_in[0];
  const float* emb  = (const float*)d_in[1];
  const float* pos  = (const float*)d_in[2];
  const float* Wq   = (const float*)d_in[3];
  const float* bq   = (const float*)d_in[4];
  const float* Wk   = (const float*)d_in[5];
  const float* bk   = (const float*)d_in[6];
  const float* Wv   = (const float*)d_in[7];
  const float* bv   = (const float*)d_in[8];
  const float* Wo   = (const float*)d_in[9];
  const float* bo   = (const float*)d_in[10];
  const float* W1   = (const float*)d_in[11];
  const float* b1   = (const float*)d_in[12];
  const float* W2   = (const float*)d_in[13];
  const float* b2   = (const float*)d_in[14];
  const float* g1   = (const float*)d_in[15];
  const float* be1  = (const float*)d_in[16];
  const float* g2   = (const float*)d_in[17];
  const float* be2  = (const float*)d_in[18];
  const float* Wout = (const float*)d_in[19];
  const float* bout = (const float*)d_in[20];
  float* out = (float*)d_out;

  // ---- workspace carve-up (~303 MiB; all offsets 256B aligned) ----
  char* ws = (char*)d_ws;
  size_t off = 0;
  auto take = [&](size_t bytes) -> char* {
    char* p = ws + off;
    off += (bytes + 255) & ~(size_t)255;
    return p;
  };
  float*  xf   = (float*)take((size_t)BS_ * D_ * 4);
  __bf16* xb   = (__bf16*)take((size_t)BS_ * D_ * 2);
  __bf16* qbuf = (__bf16*)take((size_t)BS_ * D_ * 2);
  __bf16* kbuf = (__bf16*)take((size_t)BS_ * D_ * 2);
  __bf16* vbuf = (__bf16*)take((size_t)BS_ * D_ * 2);
  __bf16* obuf = (__bf16*)take((size_t)BS_ * D_ * 2);
  float*  tmp  = (float*)take((size_t)BS_ * D_ * 4);
  float*  lnf  = (float*)take((size_t)BS_ * D_ * 4);
  __bf16* lnb  = (__bf16*)take((size_t)BS_ * D_ * 2);
  __bf16* ffb  = (__bf16*)take((size_t)BS_ * F_ * 2);
  __bf16* sbf  = (__bf16*)take((size_t)B_ * H_ * S_ * S_ * 2);
  __bf16* abf  = (__bf16*)take((size_t)B_ * H_ * S_ * S_ * 2);
  __bf16* wqkv = (__bf16*)take((size_t)L_ * 3 * D_ * D_ * 2);
  __bf16* wob  = (__bf16*)take((size_t)L_ * D_ * D_ * 2);
  __bf16* w1b  = (__bf16*)take((size_t)L_ * D_ * F_ * 2);
  __bf16* w2b  = (__bf16*)take((size_t)L_ * F_ * D_ * 2);
  __bf16* woutb= (__bf16*)take((size_t)D_ * V_ * 2);

  auto gemm = [&](const __bf16* A, long long sA1, long long sA2, int lda,
                  const __bf16* Bm, long long sB1, long long sB2, int ldb,
                  float* Cf, __bf16* Cb, long long sC1, long long sC2, int ldc,
                  const float* bias, const float* resid, long long sR1, long long sR2,
                  int M, int N, int K, int zdiv, int nz, int flags) {
    GemmArgs ga{A, sA1, sA2, lda, Bm, sB1, sB2, ldb, Cf, Cb, sC1, sC2, ldc,
                bias, resid, sR1, sR2, M, N, K, zdiv};
    launch_gemm(stream, ga, nz, flags);
  };

  // ---- weight precision conversion / repacking (per launch; deterministic) ----
  for (int l = 0; l < L_; ++l) {
    pack_qkv_kernel<<<(H_ * D_ * HD_) / 256, 256, 0, stream>>>(
        Wq + (size_t)l * H_ * D_ * HD_, wqkv + ((size_t)l * 3 + 0) * D_ * D_);
    pack_qkv_kernel<<<(H_ * D_ * HD_) / 256, 256, 0, stream>>>(
        Wk + (size_t)l * H_ * D_ * HD_, wqkv + ((size_t)l * 3 + 1) * D_ * D_);
    pack_qkv_kernel<<<(H_ * D_ * HD_) / 256, 256, 0, stream>>>(
        Wv + (size_t)l * H_ * D_ * HD_, wqkv + ((size_t)l * 3 + 2) * D_ * D_);
  }
  cvt_bf16_kernel<<<(L_ * D_ * D_) / 256, 256, 0, stream>>>(Wo, wob, (long long)L_ * D_ * D_);
  cvt_bf16_kernel<<<(L_ * D_ * F_) / 256, 256, 0, stream>>>(W1, w1b, (long long)L_ * D_ * F_);
  cvt_bf16_kernel<<<(L_ * F_ * D_) / 256, 256, 0, stream>>>(W2, w2b, (long long)L_ * F_ * D_);
  cvt_bf16_kernel<<<((size_t)D_ * V_) / 256, 256, 0, stream>>>(Wout, woutb, (long long)D_ * V_);

  // ---- embedding ----
  embed_kernel<<<(BS_ * D_) / 256, 256, 0, stream>>>(tokens, emb, pos, xf, xb);

  const long long SD = (long long)S_ * D_;
  const long long SS = (long long)S_ * S_;

  for (int l = 0; l < L_; ++l) {
    const __bf16* wq = wqkv + ((size_t)l * 3 + 0) * D_ * D_;
    const __bf16* wk = wqkv + ((size_t)l * 3 + 1) * D_ * D_;
    const __bf16* wv = wqkv + ((size_t)l * 3 + 2) * D_ * D_;

    // QKV projections (bias fused; bf16 out feeds attention WMMAs)
    gemm(xb, 0, 0, D_, wq, 0, 0, D_, nullptr, qbuf, 0, 0, D_,
         bq + l * H_ * HD_, nullptr, 0, 0, BS_, D_, D_, 1, 1, GF_BIAS | GF_WBF16);
    gemm(xb, 0, 0, D_, wk, 0, 0, D_, nullptr, kbuf, 0, 0, D_,
         bk + l * H_ * HD_, nullptr, 0, 0, BS_, D_, D_, 1, 1, GF_BIAS | GF_WBF16);
    gemm(xb, 0, 0, D_, wv, 0, 0, D_, nullptr, vbuf, 0, 0, D_,
         bv + l * H_ * HD_, nullptr, 0, 0, BS_, D_, D_, 1, 1, GF_BIAS | GF_WBF16);

    // scores[b,h] = Q[b,h] (S x 64) * K[b,h]^T -> bf16 (fits in 192MB L2)
    gemm(qbuf, SD, HD_, D_, kbuf, SD, HD_, D_,
         nullptr, sbf, (long long)H_ * SS, SS, S_,
         nullptr, nullptr, 0, 0, S_, S_, HD_, H_, B_ * H_, GF_WBF16 | GF_BTRANS);

    softmax_bf16_kernel<<<B_ * H_ * S_, 256, 0, stream>>>(sbf, abf, 0.125f);

    // O[b,h] = attn (S x S) * V[b,h] (S x 64), head-concatenated into [BS, D]
    gemm(abf, (long long)H_ * SS, SS, S_, vbuf, SD, HD_, D_,
         nullptr, obuf, SD, HD_, D_, nullptr, nullptr, 0, 0,
         S_, HD_, S_, H_, B_ * H_, GF_WBF16);

    // Wo projection + bias + residual(x) -> tmp f32
    gemm(obuf, 0, 0, D_, wob + (size_t)l * D_ * D_, 0, 0, D_,
         tmp, nullptr, 0, 0, D_, bo + l * D_, xf, 0, 0,
         BS_, D_, D_, 1, 1, GF_BIAS | GF_RESID | GF_WF32);
    layernorm_kernel<<<BS_, 256, 0, stream>>>(tmp, g1 + l * D_, be1 + l * D_, lnf, lnb);

    // FFN: relu(ln @ W1 + b1) @ W2 + b2 + residual(ln)
    gemm(lnb, 0, 0, D_, w1b + (size_t)l * D_ * F_, 0, 0, F_,
         nullptr, ffb, 0, 0, F_, b1 + l * F_, nullptr, 0, 0,
         BS_, F_, D_, 1, 1, GF_BIAS | GF_RELU | GF_WBF16);
    gemm(ffb, 0, 0, F_, w2b + (size_t)l * F_ * D_, 0, 0, D_,
         tmp, nullptr, 0, 0, D_, b2 + l * D_, lnf, 0, 0,
         BS_, D_, F_, 1, 1, GF_BIAS | GF_RESID | GF_WF32);
    layernorm_kernel<<<BS_, 256, 0, stream>>>(tmp, g2 + l * D_, be2 + l * D_, xf, xb);
  }

  // ---- output projection: [BS,1024] x [1024,32000] + bout -> logits f32 ----
  gemm(xb, 0, 0, D_, woutb, 0, 0, V_, out, nullptr, 0, 0, V_,
       bout, nullptr, 0, 0, BS_, V_, D_, 1, 1, GF_BIAS | GF_WF32);
}
